// Martinez_32341103739197
// MI455X (gfx1250) — matrix operand order repaired and checked by reading, not compiled, not run on hardware
//
#include <hip/hip_runtime.h>

#define H_    1024
#define BATCH 256
#define D_    54
#define DP    64          // D padded to multiple of 32 for WMMA K
#define SEQ_  100
#define PRED_ 50
#define G4    (4 * H_)

typedef _Float16 h16;
typedef _Float16 v16h __attribute__((ext_vector_type(16)));
typedef _Float16 v8h  __attribute__((ext_vector_type(8)));
typedef float    v8f  __attribute__((ext_vector_type(8)));
typedef int      i4v  __attribute__((ext_vector_type(4)));

// Branch-free activations: v_exp_f32 / v_rcp_f32, no libm slow path.
__device__ __forceinline__ float sigf(float x) {
    return __builtin_amdgcn_rcpf(1.0f + __expf(-x));
}
__device__ __forceinline__ float tanh_fast(float x) {
    return 2.0f * sigf(2.0f * x) - 1.0f;
}

// ---- async global->LDS (gfx1250) with sync fallback -------------------------
#if defined(__has_builtin)
#if __has_builtin(__builtin_amdgcn_global_load_async_to_lds_b128) && \
    __has_builtin(__builtin_amdgcn_s_wait_asynccnt)
#define HAS_ASYNC_LDS 1
#endif
#endif
#ifndef HAS_ASYNC_LDS
#define HAS_ASYNC_LDS 0
#endif

__device__ __forceinline__ void cp16_to_lds(const h16* g, h16* l) {
#if HAS_ASYNC_LDS
    // builtin signature (probe-confirmed): (global int4* src, lds int4* dst, imm offset, imm cpol)
    __builtin_amdgcn_global_load_async_to_lds_b128(
        (__attribute__((address_space(1))) i4v*)(h16*)g,
        (__attribute__((address_space(3))) i4v*)l, 0, 0);
#else
    *(v8h*)l = *(const v8h*)g;
#endif
}
#if HAS_ASYNC_LDS
#define ASYNC_WAIT(n) __builtin_amdgcn_s_wait_asynccnt(n)
#else
#define ASYNC_WAIT(n) ((void)0)
#endif

// ---- WMMA fragment loaders (gfx1250 16-bit layouts, wave32) -----------------
// A 16x32 f16 per lane: 8 contiguous halves at k0=half*8, 8 more at 16+half*8.
__device__ __forceinline__ v16h load_fragA(const h16* lds, int row, int half) {
    const v8h lo = *(const v8h*)(lds + row * 32 + half * 8);
    const v8h hi = *(const v8h*)(lds + row * 32 + 16 + half * 8);
    v16h a;
#pragma unroll
    for (int i = 0; i < 8; ++i) { a[i] = lo[i]; a[8 + i] = hi[i]; }
    return a;
}
// B 32x16 f16 (KxN): per lane 8 halves at k0=half*8, 8 more at 16+half*8 (N=lane&15).
// ldsB tile stored [gate][N(32)][K(32)] -> same k order as A fragment.
__device__ __forceinline__ v16h load_fragB(const h16* lds, int gate, int n, int half) {
    const v8h* p = (const v8h*)(lds + (gate * 32 + n) * 32 + half * 8);
    const v8h lo = p[0];
    const v8h hi = p[2];
    v16h b;
#pragma unroll
    for (int i = 0; i < 8; ++i) { b[i] = lo[i]; b[8 + i] = hi[i]; }
    return b;
}

// ---- Fused LSTM cell: G = x*Wih^T + h*Whh^T + b, then gate math -------------
// Grid: 64 blocks x 256 threads. Block tile: M128 x N32 x 4 gates.
// Waves 4(M)x2(N): wave tile M32 x N16 x 4 gates = 8 accumulators, 8 WMMA/chunk.
__global__ __launch_bounds__(256)
void lstm_cell_kernel(const h16* __restrict__ Ax, int lda, int Kx,   // x [B][Kx], row stride lda
                      const h16* __restrict__ Wih,                   // [4H][Kx] (K contiguous)
                      const h16* __restrict__ Ah,                    // h_prev [B][H] f16
                      const h16* __restrict__ Whh,                   // [4H][H]
                      const float* __restrict__ bias,                // [4H] = bih + bhh
                      float* __restrict__ c,                         // [B][H] in place
                      h16* __restrict__ hOut)                        // [B][H] f16
{
    __shared__ alignas(16) h16 ldsA[2][128 * 32];      // 2 x 8 KB
    __shared__ alignas(16) h16 ldsB[2][4 * 32 * 32];   // 2 x 8 KB
    const int tid   = threadIdx.x;
    const int wave  = tid >> 5;
    const int lane  = tid & 31;
    const int half  = lane >> 4;
    const int lm    = lane & 15;
    const int nbase = (blockIdx.x & 31) * 32;          // H/32 = 32 column tiles
    const int mwg   = (blockIdx.x >> 5) * 128;         // B/128 = 2 row groups
    const int mTile = (wave >> 1) * 32;                // wave row offset in block tile
    const int nTile = (wave & 1) * 16;                 // wave col offset in block tile

    // cooperative-fill mappings (each thread: 2x16B for A, 2x16B for B)
    const int rA = tid >> 1;                           // A row 0..127
    const int kA = (tid & 1) * 16;                     // A k offset (halves)
    const int gB = tid >> 6;                           // gate 0..3
    const int nB = (tid >> 1) & 31;                    // n 0..31
    const int kB = (tid & 1) * 16;                     // k offset (halves)

    const int nc0   = Kx >> 5;
    const int nctot = nc0 + (H_ >> 5);

    v8f acc[2][4] = {};                                // [m-subtile][gate]

    auto issue_tile = [&](int buf, int i) {
        const h16* A; int la; const h16* W; int Kp; int kc;
        if (i < nc0) { A = Ax; la = lda; W = Wih; Kp = Kx; kc = i << 5; }
        else         { A = Ah; la = H_;  W = Whh; Kp = H_; kc = (i - nc0) << 5; }
        const h16* sa = A + (mwg + rA) * la + kc + kA;
        h16*       da = &ldsA[buf][rA * 32 + kA];
        const h16* sb = W + (gB * H_ + nbase + nB) * Kp + kc + kB;
        h16*       db = &ldsB[buf][(gB * 32 + nB) * 32 + kB];
        cp16_to_lds(sa,     da);
        cp16_to_lds(sa + 8, da + 8);
        cp16_to_lds(sb,     db);
        cp16_to_lds(sb + 8, db + 8);
#if !HAS_ASYNC_LDS
        __builtin_prefetch(sb + 64, 0, 1);   // global_prefetch_b8 on the weight stream
#endif
    };

    issue_tile(0, 0);                                  // prologue: chunk 0 -> buf 0
#pragma unroll 1
    for (int i = 0; i < nctot; ++i) {
        const int p = i & 1;
        if (i + 1 < nctot) {
            issue_tile(1 - p, i + 1);                  // overlap next chunk's fill
            ASYNC_WAIT(4);                             // oldest 4 (buf p) complete
        } else {
            ASYNC_WAIT(0);
        }
        __syncthreads();                               // buf p visible to all waves
        const v16h a0 = load_fragA(&ldsA[p][0], mTile + lm,      half);
        const v16h a1 = load_fragA(&ldsA[p][0], mTile + 16 + lm, half);
#pragma unroll
        for (int g = 0; g < 4; ++g) {
            const v16h b = load_fragB(&ldsB[p][0], g, nTile + lm, half);
            acc[0][g] = __builtin_amdgcn_wmma_f32_16x16x32_f16(
                false, a0, false, b, (short)0, acc[0][g], false, false);
            acc[1][g] = __builtin_amdgcn_wmma_f32_16x16x32_f16(
                false, a1, false, b, (short)0, acc[1][g], false, false);
        }
        __syncthreads();                               // reads done before buf p refilled
    }

    // epilogue: C layout -> VGPR r holds row (r + 8*half), col = lane&15
    const int hcol = nbase + nTile + lm;
    const float bi = bias[hcol];
    const float bf = bias[H_ + hcol];
    const float bg = bias[2 * H_ + hcol];
    const float bo = bias[3 * H_ + hcol];
#pragma unroll
    for (int mi = 0; mi < 2; ++mi) {
#pragma unroll
        for (int r = 0; r < 8; ++r) {
            const int row = mwg + mTile + mi * 16 + r + half * 8;
            const int idx = row * H_ + hcol;
            const float iv = sigf(acc[mi][0][r] + bi);
            const float fv = sigf(acc[mi][1][r] + bf);
            const float gv = tanh_fast(acc[mi][2][r] + bg);
            const float ov = sigf(acc[mi][3][r] + bo);
            const float cn = fv * c[idx] + iv * gv;
            c[idx]    = cn;
            hOut[idx] = (h16)(ov * tanh_fast(cn));
        }
    }
}

// ---- decoder fc_in: xn = frame @ W^T + b  (K=54, tiny) ----------------------
__global__ __launch_bounds__(256)
void fc_in_kernel(const float* __restrict__ frame, const float* __restrict__ W,
                  const float* __restrict__ b, h16* __restrict__ out)
{
    const int idx  = blockIdx.x * 256 + threadIdx.x;   // B*H threads
    const int bi   = idx >> 10;
    const int hcol = idx & (H_ - 1);
    float s = b[hcol];
    const float* fr = frame + bi * D_;
    const float* wr = W + hcol * D_;
#pragma unroll
    for (int k = 0; k < D_; ++k) s += fr[k] * wr[k];
    out[idx] = (h16)s;
}

// ---- decoder fc_out: frame' = h @ W^T + b + frame; writes output ------------
// One block per batch row; reduction parallelized over 4 k-stripes of 256.
__global__ __launch_bounds__(256)
void fc_out_kernel(const h16* __restrict__ hIn, const float* __restrict__ W,
                   const float* __restrict__ b, const float* __restrict__ framePrev,
                   float* __restrict__ frameNext, float* __restrict__ out, int t)
{
    __shared__ float sh[H_];
    __shared__ float red[256];
    const int bi = blockIdx.x;
    for (int k = threadIdx.x; k < H_; k += 256) sh[k] = (float)hIn[bi * H_ + k];
    __syncthreads();
    const int q = threadIdx.x & 3;                     // k-stripe
    const int d = threadIdx.x >> 2;                    // 0..63 (active < 54)
    float s = 0.f;
    if (d < D_) {
        const float* wr = W + d * H_ + q * 256;
        const float* hr = sh + q * 256;
        for (int k = 0; k < 256; ++k) s += hr[k] * wr[k];
    }
    red[threadIdx.x] = s;
    __syncthreads();
    if (q == 0 && d < D_) {
        const float tot = red[d * 4] + red[d * 4 + 1] + red[d * 4 + 2] + red[d * 4 + 3]
                        + b[d] + framePrev[bi * D_ + d];
        frameNext[bi * D_ + d] = tot;
        out[(bi * SEQ_ + PRED_ + t) * D_ + d] = tot;
    }
}

// ---- prep kernels -----------------------------------------------------------
__global__ void cvt_f16_kernel(h16* __restrict__ dst, const float* __restrict__ src, int n) {
    const int i = blockIdx.x * 256 + threadIdx.x;
    if (i < n) dst[i] = (h16)src[i];
}
__global__ void pad_wih0_kernel(h16* __restrict__ dst, const float* __restrict__ src) {
    const int i = blockIdx.x * 256 + threadIdx.x;      // G4*DP
    const int g = i >> 6, k = i & 63;
    dst[i] = (k < D_) ? (h16)src[g * D_ + k] : (h16)0.f;
}
// fp16-pad x[:, :PRED, :] -> [B][PRED][DP]; copy prefix into out; init frame0
__global__ void prep_x_kernel(const float* __restrict__ x, h16* __restrict__ x16,
                              float* __restrict__ out, float* __restrict__ frame0) {
    const int i = blockIdx.x * 256 + threadIdx.x;      // BATCH*PRED*DP
    const int k = i & 63;
    const int t = (i >> 6) % PRED_;
    const int b = i / (PRED_ * DP);
    h16 v16 = (h16)0.f;
    if (k < D_) {
        const float v = x[(b * SEQ_ + t) * D_ + k];
        v16 = (h16)v;
        out[(b * SEQ_ + t) * D_ + k] = v;
        if (t == PRED_ - 1) frame0[b * D_ + k] = v;
    }
    x16[(b * PRED_ + t) * DP + k] = v16;
}
__global__ void vecadd_kernel(float* __restrict__ d, const float* __restrict__ a,
                              const float* __restrict__ b, int n) {
    const int i = blockIdx.x * 256 + threadIdx.x;
    if (i < n) d[i] = a[i] + b[i];
}
__global__ void fill_zero_kernel(unsigned* __restrict__ p, int n) {
    const int i = blockIdx.x * 256 + threadIdx.x;
    if (i < n) p[i] = 0u;
}

// ---- host -------------------------------------------------------------------
extern "C" void kernel_launch(void* const* d_in, const int* in_sizes, int n_in,
                              void* d_out, int out_size, void* d_ws, size_t ws_size,
                              hipStream_t stream) {
    (void)in_sizes; (void)n_in; (void)out_size; (void)ws_size;
    const float* x       = (const float*)d_in[0];
    const float* encWih0 = (const float*)d_in[1];
    const float* encWih1 = (const float*)d_in[2];
    const float* encWhh  = (const float*)d_in[3];
    const float* enc_bih = (const float*)d_in[4];
    const float* enc_bhh = (const float*)d_in[5];
    const float* decWih  = (const float*)d_in[6];
    const float* decWhh  = (const float*)d_in[7];
    const float* dec_bih = (const float*)d_in[8];
    const float* dec_bhh = (const float*)d_in[9];
    const float* fcinW   = (const float*)d_in[10];
    const float* fcinB   = (const float*)d_in[11];
    const float* fcoutW  = (const float*)d_in[12];
    const float* fcoutB  = (const float*)d_in[13];
    float* out = (float*)d_out;

    char* ws = (char*)d_ws;
    size_t off = 0;
    auto alloc = [&](size_t bytes) -> void* {
        void* p = ws + off;
        off = (off + bytes + 255) & ~(size_t)255;
        return p;
    };
    const size_t WB = (size_t)G4 * H_ * sizeof(h16);   // 8 MB per big weight
    h16* wEncWih0p = (h16*)alloc((size_t)G4 * DP * sizeof(h16));
    h16* wEncWih1  = (h16*)alloc(WB);
    h16* wEncWhh0  = (h16*)alloc(WB);
    h16* wEncWhh1  = (h16*)alloc(WB);
    h16* wDecWih0  = (h16*)alloc(WB);
    h16* wDecWih1  = (h16*)alloc(WB);
    h16* wDecWhh0  = (h16*)alloc(WB);
    h16* wDecWhh1  = (h16*)alloc(WB);
    h16* x16       = (h16*)alloc((size_t)BATCH * PRED_ * DP * sizeof(h16));
    h16* hA0 = (h16*)alloc((size_t)BATCH * H_ * sizeof(h16));
    h16* hB0 = (h16*)alloc((size_t)BATCH * H_ * sizeof(h16));
    h16* hA1 = (h16*)alloc((size_t)BATCH * H_ * sizeof(h16));
    h16* hB1 = (h16*)alloc((size_t)BATCH * H_ * sizeof(h16));
    float* c0 = (float*)alloc((size_t)BATCH * H_ * sizeof(float));
    float* c1 = (float*)alloc((size_t)BATCH * H_ * sizeof(float));
    h16* xn16 = (h16*)alloc((size_t)BATCH * H_ * sizeof(h16));
    float* bEnc0 = (float*)alloc(G4 * sizeof(float));
    float* bEnc1 = (float*)alloc(G4 * sizeof(float));
    float* bDec0 = (float*)alloc(G4 * sizeof(float));
    float* bDec1 = (float*)alloc(G4 * sizeof(float));
    float* frA = (float*)alloc((size_t)BATCH * D_ * sizeof(float));
    float* frB = (float*)alloc((size_t)BATCH * D_ * sizeof(float));

    const int nW = G4 * H_;
    const int gW = (nW + 255) / 256;
    cvt_f16_kernel<<<gW, 256, 0, stream>>>(wEncWih1, encWih1, nW);
    cvt_f16_kernel<<<gW, 256, 0, stream>>>(wEncWhh0, encWhh,      nW);
    cvt_f16_kernel<<<gW, 256, 0, stream>>>(wEncWhh1, encWhh + nW, nW);
    cvt_f16_kernel<<<gW, 256, 0, stream>>>(wDecWih0, decWih,      nW);
    cvt_f16_kernel<<<gW, 256, 0, stream>>>(wDecWih1, decWih + nW, nW);
    cvt_f16_kernel<<<gW, 256, 0, stream>>>(wDecWhh0, decWhh,      nW);
    cvt_f16_kernel<<<gW, 256, 0, stream>>>(wDecWhh1, decWhh + nW, nW);
    pad_wih0_kernel<<<(G4 * DP) / 256, 256, 0, stream>>>(wEncWih0p, encWih0);
    prep_x_kernel<<<(BATCH * PRED_ * DP) / 256, 256, 0, stream>>>(x, x16, out, frA);
    vecadd_kernel<<<G4 / 256, 256, 0, stream>>>(bEnc0, enc_bih,      enc_bhh,      G4);
    vecadd_kernel<<<G4 / 256, 256, 0, stream>>>(bEnc1, enc_bih + G4, enc_bhh + G4, G4);
    vecadd_kernel<<<G4 / 256, 256, 0, stream>>>(bDec0, dec_bih,      dec_bhh,      G4);
    vecadd_kernel<<<G4 / 256, 256, 0, stream>>>(bDec1, dec_bih + G4, dec_bhh + G4, G4);
    const int nZ16 = BATCH * H_ / 2;
    fill_zero_kernel<<<(nZ16 + 255) / 256, 256, 0, stream>>>((unsigned*)hA0, nZ16);
    fill_zero_kernel<<<(nZ16 + 255) / 256, 256, 0, stream>>>((unsigned*)hA1, nZ16);
    const int nZ32 = BATCH * H_;
    fill_zero_kernel<<<(nZ32 + 255) / 256, 256, 0, stream>>>((unsigned*)c0, nZ32);
    fill_zero_kernel<<<(nZ32 + 255) / 256, 256, 0, stream>>>((unsigned*)c1, nZ32);

    h16 *h0p = hA0, *h0n = hB0, *h1p = hA1, *h1n = hB1;
    // ---- encoder: 50 steps x 2 layers ----
    for (int t = 0; t < PRED_; ++t) {
        lstm_cell_kernel<<<64, 256, 0, stream>>>(x16 + t * DP, PRED_ * DP, DP,
                                                 wEncWih0p, h0p, wEncWhh0, bEnc0, c0, h0n);
        lstm_cell_kernel<<<64, 256, 0, stream>>>(h0n, H_, H_,
                                                 wEncWih1, h1p, wEncWhh1, bEnc1, c1, h1n);
        h16* s;
        s = h0p; h0p = h0n; h0n = s;
        s = h1p; h1p = h1n; h1n = s;
    }
    // ---- decoder: 50 autoregressive steps ----
    float *fcur = frA, *fnext = frB;
    for (int t = 0; t < SEQ_ - PRED_; ++t) {
        fc_in_kernel<<<BATCH * H_ / 256, 256, 0, stream>>>(fcur, fcinW, fcinB, xn16);
        lstm_cell_kernel<<<64, 256, 0, stream>>>(xn16, H_, H_,
                                                 wDecWih0, h0p, wDecWhh0, bDec0, c0, h0n);
        lstm_cell_kernel<<<64, 256, 0, stream>>>(h0n, H_, H_,
                                                 wDecWih1, h1p, wDecWhh1, bDec1, c1, h1n);
        fc_out_kernel<<<256, 256, 0, stream>>>(h1n, fcoutW, fcoutB, fcur, fnext, out, t);
        h16* s;
        s = h0p; h0p = h0n; h0n = s;
        s = h1p; h1p = h1n; h1n = s;
        float* f = fcur; fcur = fnext; fnext = f;
    }
}
